// DualBranchPredictor_86174223827124
// MI455X (gfx1250) — compile-verified, matrix operand order checked by reading, not run on hardware
//
#include <hip/hip_runtime.h>
#include <hip/hip_bf16.h>

// ---------------- problem constants ----------------
#define NNODES   28160      // B * (NP + NL)
#define NEDGES   450560     // NNODES * 16
#define ETOT     478720     // NEDGES + NNODES self loops
#define HID      256
#define HGAT     4
#define NLAYERS  3
#define NGRAPH   64
#define NPG      440
#define NPROT    400
#define KXPAD    1312       // 1310 padded to multiple of 32
#define KLPAD    64         // 36 padded

typedef __attribute__((ext_vector_type(16))) __bf16 v16bf;
typedef __attribute__((ext_vector_type(8)))  __bf16 v8bf;
typedef __attribute__((ext_vector_type(8)))  float  v8f;

#define WMMA_BF16(a, b, c) \
  __builtin_amdgcn_wmma_f32_16x16x32_bf16(false, (a), false, (b), (short)0, (c), false, false)

// ---------------- helpers ----------------
__device__ __forceinline__ float blk_sum256(float v, float* red) {
  int t = threadIdx.x;
  red[t] = v; __syncthreads();
  for (int s = 128; s > 0; s >>= 1) { if (t < s) red[t] += red[t + s]; __syncthreads(); }
  float r = red[0]; __syncthreads();
  return r;
}

__device__ __forceinline__ v16bf frag_ld(const __bf16* p) {
  v8bf lo = *(const v8bf*)(p);
  v8bf hi = *(const v8bf*)(p + 16);
  v16bf f;
#pragma unroll
  for (int i = 0; i < 8; ++i) { f[i] = lo[i]; f[8 + i] = hi[i]; }
  return f;
}

// ---------------- conversion: f32 -> bf16 with K padding ----------------
__global__ void cvt_pad_bf16(const float* __restrict__ src, __hip_bfloat16* __restrict__ dst,
                             int K, int ldd, long total) {
  long i = (long)blockIdx.x * blockDim.x + threadIdx.x;
  if (i >= total) return;
  long r = i / ldd; int c = (int)(i - r * ldd);
  float v = (c < K) ? src[r * (long)K + c] : 0.0f;
  dst[i] = __float2bfloat16(v);
}

// ---------------- bf16 WMMA GEMM: C[M,N] = A[M,K] * B[N,K]^T ----------------
// one wave computes a 32x64 macro-tile (2 M-tiles x 4 N-tiles, 8 accumulators).
// software-pipelined: next K-step's 6 fragments are in flight while the current
// 8 v_wmma ops execute; launch_bounds lifts the VGPR cap so both fragment sets
// stay live (no full s_wait_loadcnt drains inside the loop).
__global__ void __launch_bounds__(256, 1)
wmma_gemm_bf16(const __bf16* __restrict__ A, int lda,
               const __bf16* __restrict__ B, int ldb,
               float* __restrict__ C, int ldc,
               int ngroups, int ksteps, int totalWaves) {
  int wave = threadIdx.x >> 5;
  int lane = threadIdx.x & 31;
  int w = blockIdx.x * (blockDim.x >> 5) + wave;
  if (w >= totalWaves) return;                    // wave-uniform; EXEC stays all-ones
  int gm = w / ngroups;
  int gn = w - gm * ngroups;
  int krow = (lane >> 4) << 3;                    // 0 or 8 : K sub-block per half-wave
  const __bf16* ap = A + (size_t)(gm * 32 + (lane & 15)) * lda + krow;
  const __bf16* bp = B + (size_t)(gn * 64 + (lane & 15)) * ldb + krow;
  const size_t astep = (size_t)16 * lda;          // next M-tile (rows +16)
  const size_t bstep = (size_t)16 * ldb;          // next N-tile (rows +16)

  v16bf a0 = frag_ld(ap);
  v16bf a1 = frag_ld(ap + astep);
  v16bf b0 = frag_ld(bp);
  v16bf b1 = frag_ld(bp + bstep);
  v16bf b2 = frag_ld(bp + 2 * bstep);
  v16bf b3 = frag_ld(bp + 3 * bstep);

  v8f acc[2][4] = {};
  for (int k = 0; k < ksteps - 1; ++k) {
    ap += 32; bp += 32;
    v16bf na0 = frag_ld(ap);
    v16bf na1 = frag_ld(ap + astep);
    v16bf nb0 = frag_ld(bp);
    v16bf nb1 = frag_ld(bp + bstep);
    v16bf nb2 = frag_ld(bp + 2 * bstep);
    v16bf nb3 = frag_ld(bp + 3 * bstep);
    acc[0][0] = WMMA_BF16(a0, b0, acc[0][0]);
    acc[0][1] = WMMA_BF16(a0, b1, acc[0][1]);
    acc[0][2] = WMMA_BF16(a0, b2, acc[0][2]);
    acc[0][3] = WMMA_BF16(a0, b3, acc[0][3]);
    acc[1][0] = WMMA_BF16(a1, b0, acc[1][0]);
    acc[1][1] = WMMA_BF16(a1, b1, acc[1][1]);
    acc[1][2] = WMMA_BF16(a1, b2, acc[1][2]);
    acc[1][3] = WMMA_BF16(a1, b3, acc[1][3]);
    a0 = na0; a1 = na1; b0 = nb0; b1 = nb1; b2 = nb2; b3 = nb3;
  }
  acc[0][0] = WMMA_BF16(a0, b0, acc[0][0]);
  acc[0][1] = WMMA_BF16(a0, b1, acc[0][1]);
  acc[0][2] = WMMA_BF16(a0, b2, acc[0][2]);
  acc[0][3] = WMMA_BF16(a0, b3, acc[0][3]);
  acc[1][0] = WMMA_BF16(a1, b0, acc[1][0]);
  acc[1][1] = WMMA_BF16(a1, b1, acc[1][1]);
  acc[1][2] = WMMA_BF16(a1, b2, acc[1][2]);
  acc[1][3] = WMMA_BF16(a1, b3, acc[1][3]);

#pragma unroll
  for (int mi = 0; mi < 2; ++mi) {
#pragma unroll
    for (int ni = 0; ni < 4; ++ni) {
      float* crow = C + (size_t)(gm * 32 + mi * 16 + (lane >> 4) * 8) * ldc
                      + gn * 64 + ni * 16 + (lane & 15);
#pragma unroll
      for (int r = 0; r < 8; ++r) crow[(size_t)r * ldc] = acc[mi][ni][r];
    }
  }
}

// ---------------- input projection epilogue: LN + relu + select + nt_emb ----------------
__global__ void ln_select_kernel(const float* __restrict__ hpre, const float* __restrict__ hlpre,
                                 const float* __restrict__ prot_b, const float* __restrict__ prot_g,
                                 const float* __restrict__ prot_beta,
                                 const float* __restrict__ lig_b, const float* __restrict__ lig_g,
                                 const float* __restrict__ lig_beta,
                                 const int* __restrict__ node_type, const float* __restrict__ nt_emb,
                                 float* __restrict__ h, __hip_bfloat16* __restrict__ hbf) {
  int n = blockIdx.x, c = threadIdx.x;
  __shared__ float red[256];
  float vp = hpre[(size_t)n * HID + c] + prot_b[c];
  float vl = hlpre[(size_t)n * HID + c] + lig_b[c];
  float mp = blk_sum256(vp, red) * (1.0f / HID);
  float dp = vp - mp;
  float varp = blk_sum256(dp * dp, red) * (1.0f / HID);
  float hp = fmaxf(dp * rsqrtf(varp + 1e-5f) * prot_g[c] + prot_beta[c], 0.0f);
  float ml = blk_sum256(vl, red) * (1.0f / HID);
  float dl = vl - ml;
  float varl = blk_sum256(dl * dl, red) * (1.0f / HID);
  float hl = fmaxf(dl * rsqrtf(varl + 1e-5f) * lig_g[c] + lig_beta[c], 0.0f);
  int t = node_type[n];
  float val = ((t == 0) ? hp : hl) + nt_emb[(size_t)t * HID + c];
  h[(size_t)n * HID + c] = val;
  hbf[(size_t)n * HID + c] = __float2bfloat16(val);
}

// ---------------- per-node attention coefficients a_src / a_dst ----------------
__global__ void attn_coeff_kernel(const float* __restrict__ hp4,
                                  const float* __restrict__ asrc, const float* __restrict__ adst,
                                  float* __restrict__ a_src, float* __restrict__ a_dst) {
  int n = blockIdx.x, t = threadIdx.x;
  int head = t >> 6, j = t & 63;
  __shared__ float rs[256], rd[256];
  const float* row = hp4 + (size_t)n * (HGAT * HID) + head * HID;
  float ss = 0.0f, sd = 0.0f;
  for (int c = j; c < HID; c += 64) {
    float v = row[c];
    ss += v * asrc[head * HID + c];
    sd += v * adst[head * HID + c];
  }
  rs[t] = ss; rd[t] = sd; __syncthreads();
  for (int s = 32; s > 0; s >>= 1) {
    if (j < s) { rs[t] += rs[t + s]; rd[t] += rd[t + s]; }
    __syncthreads();
  }
  if (j == 0) { a_src[n * HGAT + head] = rs[t]; a_dst[n * HGAT + head] = rd[t]; }
}

// ---------------- edge softmax pass 1: leaky_relu logits + segment max ----------------
__global__ void edge_logits_kernel(const int* __restrict__ ei,
                                   const float* __restrict__ a_src, const float* __restrict__ a_dst,
                                   float* __restrict__ e_edge, unsigned* __restrict__ seg_max) {
  int i = blockIdx.x * 256 + threadIdx.x;            // i < ETOT (grid exact)
  int s = (i < NEDGES) ? ei[i] : (i - NEDGES);
  int d = (i < NEDGES) ? ei[NEDGES + i] : (i - NEDGES);
#pragma unroll
  for (int hh = 0; hh < HGAT; ++hh) {
    float e = a_src[s * HGAT + hh] + a_dst[d * HGAT + hh];
    e = (e > 0.0f) ? e : 0.2f * e;                   // leaky_relu(., 0.2)
    e_edge[(size_t)i * HGAT + hh] = e;
    unsigned u = __float_as_uint(e);
    unsigned k = (u & 0x80000000u) ? ~u : (u | 0x80000000u);   // order-preserving
    atomicMax(&seg_max[d * HGAT + hh], k);
  }
}

// ---------------- edge softmax pass 2: exp + segment sum ----------------
__global__ void edge_exp_kernel(const int* __restrict__ ei,
                                float* __restrict__ e_edge,
                                const unsigned* __restrict__ seg_max,
                                float* __restrict__ seg_sum) {
  int i = blockIdx.x * 256 + threadIdx.x;
  int d = (i < NEDGES) ? ei[NEDGES + i] : (i - NEDGES);
#pragma unroll
  for (int hh = 0; hh < HGAT; ++hh) {
    unsigned k = seg_max[d * HGAT + hh];
    unsigned u = (k & 0x80000000u) ? (k ^ 0x80000000u) : ~k;
    float mx = __uint_as_float(u);
    float ee = __expf(e_edge[(size_t)i * HGAT + hh] - mx);
    e_edge[(size_t)i * HGAT + hh] = ee;
    atomicAdd(&seg_sum[d * HGAT + hh], ee);
  }
}

// ---------------- edge scatter: out[dst] += alpha * hp4[src] ----------------
__global__ void edge_scatter_kernel(const int* __restrict__ ei,
                                    const float* __restrict__ e_edge,
                                    const float* __restrict__ seg_sum,
                                    const float* __restrict__ hp4,
                                    float* __restrict__ outacc) {
  int i = blockIdx.x;                                 // one block per edge
  int s = (i < NEDGES) ? ei[i] : (i - NEDGES);
  int d = (i < NEDGES) ? ei[NEDGES + i] : (i - NEDGES);
  __shared__ float alpha[HGAT];
  if (threadIdx.x < HGAT)
    alpha[threadIdx.x] = e_edge[(size_t)i * HGAT + threadIdx.x] / seg_sum[d * HGAT + threadIdx.x];
  __syncthreads();
  int c = threadIdx.x;
  const float* src = hp4 + (size_t)s * (HGAT * HID);
  float* dst = outacc + (size_t)d * (HGAT * HID);
#pragma unroll
  for (int hh = 0; hh < HGAT; ++hh)
    atomicAdd(&dst[hh * HID + c], alpha[hh] * src[hh * HID + c]);
}

// ---------------- node update: head-mean + bias + residual + LN + relu ----------------
__global__ void node_update_kernel(const float* __restrict__ outacc, const float* __restrict__ bias,
                                   const float* __restrict__ g, const float* __restrict__ b,
                                   float* __restrict__ h, __hip_bfloat16* __restrict__ hbf,
                                   int residual) {
  int n = blockIdx.x, c = threadIdx.x;
  __shared__ float red[256];
  const float* row = outacc + (size_t)n * (HGAT * HID);
  float v = 0.25f * (row[c] + row[HID + c] + row[2 * HID + c] + row[3 * HID + c]) + bias[c];
  if (residual) v += h[(size_t)n * HID + c];
  float m = blk_sum256(v, red) * (1.0f / HID);
  float dv = v - m;
  float var = blk_sum256(dv * dv, red) * (1.0f / HID);
  float o = fmaxf(dv * rsqrtf(var + 1e-5f) * g[c] + b[c], 0.0f);
  h[(size_t)n * HID + c] = o;
  hbf[(size_t)n * HID + c] = __float2bfloat16(o);
}

// ---------------- pooling scores: tanh(M@W1^T+b1)@W2^T+b2 ----------------
__global__ void scores_kernel(const float* __restrict__ h, const float* __restrict__ W1,
                              const float* __restrict__ b1, const float* __restrict__ W2,
                              const float* __restrict__ b2, float* __restrict__ scores) {
  int n = blockIdx.x, t = threadIdx.x;                // 128 threads
  __shared__ float hid[128];
  const float* row = h + (size_t)n * HID;
  float s = b1[t];
  for (int c = 0; c < HID; ++c) s += row[c] * W1[t * HID + c];
  hid[t] = tanhf(s);
  __syncthreads();
  if (t < HGAT) {
    float sc = b2[t];
    for (int j = 0; j < 128; ++j) sc += hid[j] * W2[t * 128 + j];
    scores[n * HGAT + t] = sc;
  }
}

// ---------------- per-graph pooling (graphs contiguous: 440 nodes, 400 protein) ----------------
__global__ void pool_kernel(const float* __restrict__ h, const float* __restrict__ scores,
                            float* __restrict__ combined) {
  int g = blockIdx.x, t = threadIdx.x;                // 256 threads
  __shared__ float aw[NPROT * HGAT];
  __shared__ float red[256];
  __shared__ float hmax[HGAT], hsum[HGAT];
  int head = t >> 6, j = t & 63;
  const float* sc = scores + (size_t)g * NPG * HGAT;
  float mx = -3.4e38f;
  for (int n = j; n < NPROT; n += 64) mx = fmaxf(mx, sc[n * HGAT + head]);
  red[t] = mx; __syncthreads();
  for (int s = 32; s > 0; s >>= 1) { if (j < s) red[t] = fmaxf(red[t], red[t + s]); __syncthreads(); }
  if (j == 0) hmax[head] = red[t];
  __syncthreads();
  float sm = 0.0f;
  for (int n = j; n < NPROT; n += 64) {
    float e = __expf(sc[n * HGAT + head] - hmax[head]);
    aw[n * HGAT + head] = e; sm += e;
  }
  red[t] = sm; __syncthreads();
  for (int s = 32; s > 0; s >>= 1) { if (j < s) red[t] += red[t + s]; __syncthreads(); }
  if (j == 0) hsum[head] = red[t];
  __syncthreads();
  for (int n = j; n < NPROT; n += 64) aw[n * HGAT + head] /= hsum[head];
  __syncthreads();
  const float* M = h + (size_t)g * NPG * HID;
  float a0 = 0, a1 = 0, a2 = 0, a3 = 0, gp = 0;
  for (int n = 0; n < NPROT; ++n) {
    float m = M[(size_t)n * HID + t];
    gp += m;
    a0 += aw[n * HGAT + 0] * m; a1 += aw[n * HGAT + 1] * m;
    a2 += aw[n * HGAT + 2] * m; a3 += aw[n * HGAT + 3] * m;
  }
  combined[(size_t)g * 512 + t] = 0.25f * (a0 + a1 + a2 + a3);
  combined[(size_t)g * 512 + 256 + t] = gp * (1.0f / NPG);
}

// ---------------- small dense layer ----------------
__global__ void dense_kernel(const float* __restrict__ in, const float* __restrict__ W,
                             const float* __restrict__ b, float* __restrict__ out,
                             int K, int Nout, int relu) {
  int r = blockIdx.x, t = threadIdx.x;
  if (t >= Nout) return;
  const float* row = in + (size_t)r * K;
  float s = b[t];
  for (int k = 0; k < K; ++k) s += row[k] * W[t * K + k];
  if (relu) s = fmaxf(s, 0.0f);
  out[(size_t)r * Nout + t] = s;
}

// ---------------- workspace layout ----------------
static constexpr size_t alignup(size_t x) { return (x + 255) & ~size_t(255); }

extern "C" void kernel_launch(void* const* d_in, const int* in_sizes, int n_in,
                              void* d_out, int out_size, void* d_ws, size_t ws_size,
                              hipStream_t stream) {
  (void)in_sizes; (void)n_in; (void)out_size; (void)ws_size;
  const float* x        = (const float*)d_in[0];
  const int*   ei       = (const int*)d_in[1];
  const int*   node_ty  = (const int*)d_in[3];
  const float* prot_W   = (const float*)d_in[4];
  const float* prot_b   = (const float*)d_in[5];
  const float* prot_g   = (const float*)d_in[6];
  const float* prot_be  = (const float*)d_in[7];
  const float* lig_W    = (const float*)d_in[8];
  const float* lig_b    = (const float*)d_in[9];
  const float* lig_g    = (const float*)d_in[10];
  const float* lig_be   = (const float*)d_in[11];
  const float* nt_emb   = (const float*)d_in[12];
  const float* gat_W    = (const float*)d_in[13];
  const float* gat_asrc = (const float*)d_in[14];
  const float* gat_adst = (const float*)d_in[15];
  const float* gat_bias = (const float*)d_in[16];
  const float* norm_g   = (const float*)d_in[17];
  const float* norm_b   = (const float*)d_in[18];
  const float* W1       = (const float*)d_in[19];
  const float* b1       = (const float*)d_in[20];
  const float* W2       = (const float*)d_in[21];
  const float* b2       = (const float*)d_in[22];
  const float* proj_W   = (const float*)d_in[23];
  const float* proj_b   = (const float*)d_in[24];
  const float* c1_W     = (const float*)d_in[25];
  const float* c1_b     = (const float*)d_in[26];
  const float* c2_W     = (const float*)d_in[27];
  const float* c2_b     = (const float*)d_in[28];
  const float* c3_W     = (const float*)d_in[29];
  const float* c3_b     = (const float*)d_in[30];

  char* ws = (char*)d_ws;
  size_t off = 0;
  auto carve = [&](size_t bytes) { char* p = ws + off; off = alignup(off + bytes); return p; };

  __hip_bfloat16* x_bf    = (__hip_bfloat16*)carve((size_t)NNODES * KXPAD * 2);
  __hip_bfloat16* protWbf = (__hip_bfloat16*)carve((size_t)HID * KXPAD * 2);
  __hip_bfloat16* ligWbf  = (__hip_bfloat16*)carve((size_t)HID * KLPAD * 2);
  __hip_bfloat16* gatWbf  = (__hip_bfloat16*)carve((size_t)NLAYERS * HGAT * HID * HID * 2);
  float*          h       = (float*)carve((size_t)NNODES * HID * 4);
  __hip_bfloat16* hbf     = (__hip_bfloat16*)carve((size_t)NNODES * HID * 2);
  float*          hp4     = (float*)carve((size_t)NNODES * HGAT * HID * 4);
  float*          outacc  = (float*)carve((size_t)NNODES * HGAT * HID * 4);
  float*          a_src   = (float*)carve((size_t)NNODES * HGAT * 4);
  float*          a_dst   = (float*)carve((size_t)NNODES * HGAT * 4);
  float*          e_edge  = (float*)carve((size_t)ETOT * HGAT * 4);
  unsigned*       seg_max = (unsigned*)carve((size_t)NNODES * HGAT * 4);
  float*          seg_sum = (float*)carve((size_t)NNODES * HGAT * 4);
  float*          scores  = (float*)carve((size_t)NNODES * HGAT * 4);
  float*          combined= (float*)carve((size_t)NGRAPH * 512 * 4);
  float*          g1      = (float*)carve((size_t)NGRAPH * 256 * 4);
  float*          g2      = (float*)carve((size_t)NGRAPH * 256 * 4);
  float*          g3      = (float*)carve((size_t)NGRAPH * 128 * 4);

  // hpre/hlpre alias the first/second quarters of hp4 (only live before layer 0)
  float* hpre  = hp4;
  float* hlpre = hp4 + (size_t)NNODES * HID;

  // ---- bf16 conversions (x padded to 1312, weights padded) ----
  {
    long tot = (long)NNODES * KXPAD;
    cvt_pad_bf16<<<(unsigned)((tot + 255) / 256), 256, 0, stream>>>(x, x_bf, 1310, KXPAD, tot);
  }
  {
    long tot = (long)HID * KXPAD;
    cvt_pad_bf16<<<(unsigned)((tot + 255) / 256), 256, 0, stream>>>(prot_W, protWbf, 1310, KXPAD, tot);
  }
  {
    long tot = (long)HID * KLPAD;
    cvt_pad_bf16<<<(unsigned)((tot + 255) / 256), 256, 0, stream>>>(lig_W, ligWbf, 36, KLPAD, tot);
  }
  {
    long tot = (long)NLAYERS * HGAT * HID * HID;
    cvt_pad_bf16<<<(unsigned)((tot + 255) / 256), 256, 0, stream>>>(gat_W, gatWbf, HID, HID, tot);
  }

  const int mgroups = NNODES / 32;  // 880 macro-rows of 32
  // ---- input projections (WMMA, 32x64 macro-tiles) ----
  {
    int ngroups = HID / 64;                         // 4
    int total = mgroups * ngroups;                  // 3520 waves
    wmma_gemm_bf16<<<total / 8, 256, 0, stream>>>((const __bf16*)x_bf, KXPAD,
                                                  (const __bf16*)protWbf, KXPAD,
                                                  hpre, HID, ngroups, KXPAD / 32, total);
    wmma_gemm_bf16<<<total / 8, 256, 0, stream>>>((const __bf16*)x_bf, KXPAD,
                                                  (const __bf16*)ligWbf, KLPAD,
                                                  hlpre, HID, ngroups, KLPAD / 32, total);
  }
  ln_select_kernel<<<NNODES, 256, 0, stream>>>(hpre, hlpre, prot_b, prot_g, prot_be,
                                               lig_b, lig_g, lig_be, node_ty, nt_emb, h, hbf);

  // ---- GAT layers ----
  for (int layer = 0; layer < NLAYERS; ++layer) {
    {
      int ngroups = (HGAT * HID) / 64;              // 16
      int total = mgroups * ngroups;                // 14080 waves
      wmma_gemm_bf16<<<total / 8, 256, 0, stream>>>((const __bf16*)hbf, HID,
                                                    (const __bf16*)(gatWbf + (size_t)layer * HGAT * HID * HID), HID,
                                                    hp4, HGAT * HID, ngroups, HID / 32, total);
    }
    attn_coeff_kernel<<<NNODES, 256, 0, stream>>>(hp4,
                                                  gat_asrc + (size_t)layer * HGAT * HID,
                                                  gat_adst + (size_t)layer * HGAT * HID,
                                                  a_src, a_dst);
    hipMemsetAsync(seg_max, 0, (size_t)NNODES * HGAT * 4, stream);   // 0 < encode(any finite)
    hipMemsetAsync(seg_sum, 0, (size_t)NNODES * HGAT * 4, stream);
    hipMemsetAsync(outacc, 0, (size_t)NNODES * HGAT * HID * 4, stream);
    edge_logits_kernel<<<ETOT / 256, 256, 0, stream>>>(ei, a_src, a_dst, e_edge, seg_max);
    edge_exp_kernel<<<ETOT / 256, 256, 0, stream>>>(ei, e_edge, seg_max, seg_sum);
    edge_scatter_kernel<<<ETOT, 256, 0, stream>>>(ei, e_edge, seg_sum, hp4, outacc);
    node_update_kernel<<<NNODES, 256, 0, stream>>>(outacc,
                                                   gat_bias + (size_t)layer * HID,
                                                   norm_g + (size_t)layer * HID,
                                                   norm_b + (size_t)layer * HID,
                                                   h, hbf, layer > 0 ? 1 : 0);
  }

  // ---- pooling ----
  scores_kernel<<<NNODES, 128, 0, stream>>>(h, W1, b1, W2, b2, scores);
  pool_kernel<<<NGRAPH, 256, 0, stream>>>(h, scores, combined);

  // ---- MLP head ----
  dense_kernel<<<NGRAPH, 256, 0, stream>>>(combined, proj_W, proj_b, g1, 512, 256, 1);
  dense_kernel<<<NGRAPH, 256, 0, stream>>>(g1, c1_W, c1_b, g2, 256, 256, 1);
  dense_kernel<<<NGRAPH, 128, 0, stream>>>(g2, c2_W, c2_b, g3, 256, 128, 1);
  dense_kernel<<<NGRAPH, 32, 0, stream>>>(g3, c3_W, c3_b, (float*)d_out, 128, 2, 0);
}
